// BayesianMambaBlock_58746562674867
// MI455X (gfx1250) — matrix operand order repaired and checked by reading, not compiled, not run on hardware
//
#include <hip/hip_runtime.h>
#include <hip/hip_bf16.h>

// ---------------------------------------------------------------------------
// BayesianMambaBlock on MI455X (gfx1250).
// Big GEMMs: bf16x3 split-precision WMMA (v_wmma_f32_16x16x32_bf16), with all
// fp32->bf16(hi,lo) splitting hoisted OUT of the GEMM inner loop: activations
// and weights are split exactly once, the GEMM streams bf16 through LDS.
// Param GEMM: f32 16x16x4 WMMA. Scan: register-resident recurrence.
// ---------------------------------------------------------------------------

typedef __attribute__((ext_vector_type(2)))  float  v2f;
typedef __attribute__((ext_vector_type(8)))  float  v8f;
typedef __attribute__((ext_vector_type(16))) __bf16 v16bf;
typedef unsigned short u16;

#define D_MODEL 768
#define D_INNER 1536
#define D_STATE 16
#define D_CONV  4
#define BATCH   2
#define SEQ     2048
#define NTOK    (BATCH * SEQ)   // 4096

__device__ __forceinline__ v8f wmma_f32(v2f a, v2f b, v8f c) {
  return __builtin_amdgcn_wmma_f32_16x16x4_f32(false, a, false, b, (short)0, c,
                                               false, false);
}
__device__ __forceinline__ v8f wmma_bf16(v16bf a, v16bf b, v8f c) {
  return __builtin_amdgcn_wmma_f32_16x16x32_bf16(false, a, false, b, (short)0,
                                                 c, false, false);
}

__device__ __forceinline__ float sigmoidf_(float x) {
  return 1.0f / (1.0f + expf(-x));
}
__device__ __forceinline__ float siluf_(float x) { return x * sigmoidf_(x); }

// ---- bf16 split helpers ----------------------------------------------------
__device__ __forceinline__ u16 f32_to_bf16(float f) {
  unsigned int u = __builtin_bit_cast(unsigned int, f);
  u += 0x7FFFu + ((u >> 16) & 1u);  // RNE
  return (u16)(u >> 16);
}
__device__ __forceinline__ float bf16_to_f32(u16 h) {
  unsigned int u = ((unsigned int)h) << 16;
  return __builtin_bit_cast(float, u);
}
__device__ __forceinline__ void split1(float x, u16& h, u16& l) {
  h = f32_to_bf16(x);
  l = f32_to_bf16(x - bf16_to_f32(h));
}

struct Two4 { uint4 x, y; };
// Load a 16-element bf16 fragment as two b128 LDS reads.
__device__ __forceinline__ v16bf ld_frag(const u16* p, int off2) {
  Two4 t;
  t.x = *(const uint4*)(p);
  t.y = *(const uint4*)(p + off2);
  return __builtin_bit_cast(v16bf, t);
}

// ---------------------------------------------------------------------------
// Elementwise fp32 -> (bf16 hi, bf16 lo). n must be a multiple of 1024.
// ---------------------------------------------------------------------------
__global__ __launch_bounds__(256) void split_kernel(
    const float* __restrict__ src, u16* __restrict__ hi, u16* __restrict__ lo) {
  const int i4 = (blockIdx.x * 256 + threadIdx.x) * 4;
  const float4 v = *(const float4*)(src + i4);
  u16 h0, h1, h2, h3, l0, l1, l2, l3;
  split1(v.x, h0, l0);
  split1(v.y, h1, l1);
  split1(v.z, h2, l2);
  split1(v.w, h3, l3);
  uint2 H, L;
  H.x = (unsigned)h0 | ((unsigned)h1 << 16);
  H.y = (unsigned)h2 | ((unsigned)h3 << 16);
  L.x = (unsigned)l0 | ((unsigned)l1 << 16);
  L.y = (unsigned)l2 | ((unsigned)l3 << 16);
  *(uint2*)(hi + i4) = H;
  *(uint2*)(lo + i4) = L;
}

// ---------------------------------------------------------------------------
// LayerNorm: one block per token; writes xn directly as split bf16.
// ---------------------------------------------------------------------------
__global__ __launch_bounds__(256) void layernorm_kernel(
    const float* __restrict__ x, const float* __restrict__ g,
    const float* __restrict__ b, u16* __restrict__ xnhi,
    u16* __restrict__ xnlo) {
  __shared__ float red[256];
  const int tid = threadIdx.x;
  const float* xr = x + (size_t)blockIdx.x * D_MODEL;
  float v0 = xr[tid], v1 = xr[tid + 256], v2 = xr[tid + 512];

  red[tid] = v0 + v1 + v2;
  __syncthreads();
  for (int off = 128; off > 0; off >>= 1) {
    if (tid < off) red[tid] += red[tid + off];
    __syncthreads();
  }
  const float mu = red[0] * (1.0f / (float)D_MODEL);
  __syncthreads();

  float q0 = v0 - mu, q1 = v1 - mu, q2 = v2 - mu;
  red[tid] = q0 * q0 + q1 * q1 + q2 * q2;
  __syncthreads();
  for (int off = 128; off > 0; off >>= 1) {
    if (tid < off) red[tid] += red[tid + off];
    __syncthreads();
  }
  const float rs = rsqrtf(red[0] * (1.0f / (float)D_MODEL) + 1e-5f);

  const size_t o = (size_t)blockIdx.x * D_MODEL;
#pragma unroll
  for (int i = 0; i < 3; i++) {
    const int c = tid + i * 256;
    const float q = (i == 0 ? q0 : (i == 1 ? q1 : q2));
    const float v = q * rs * g[c] + b[c];
    u16 h, l;
    split1(v, h, l);
    xnhi[o + c] = h;
    xnlo[o + c] = l;
  }
}

// ---------------------------------------------------------------------------
// bf16x3 WMMA GEMM on pre-split operands.
// out = epilogue(A @ W^T + bias).  A: [M,K] (hi/lo), W: [N,K] (hi/lo).
// Block tile 64(M) x 128(N), BK=32; 8 waves = 4M x 2N; wave tile 16x64.
// LDS tiles [row][k], row stride RS=40 bf16 (80B) -> fragments are two
// ds_load_b128 each per the ISA 16-bit A/B VGPR layouts.
// MODE 0: outF = s
// MODE 1: outF = (auxHi+auxLo) * sigmoid(s)
// MODE 2: (outHi,outLo) = split(silu(s) * auxF)
// MODE 3: outF = s + auxF
// ---------------------------------------------------------------------------
#define RS 40

template <int MODE>
__global__ __launch_bounds__(256) void gemm_bf16x3_kernel(
    const u16* __restrict__ Ahi_g, const u16* __restrict__ Alo_g,
    const u16* __restrict__ Whi_g, const u16* __restrict__ Wlo_g,
    const float* __restrict__ bias, const float* __restrict__ auxF,
    const u16* __restrict__ auxHi, const u16* __restrict__ auxLo,
    float* __restrict__ outF, u16* __restrict__ outHi, u16* __restrict__ outLo,
    int M, int N, int K) {
  __shared__ u16 Ahi[64 * RS];
  __shared__ u16 Alo[64 * RS];
  __shared__ u16 Bhi[128 * RS];
  __shared__ u16 Blo[128 * RS];

  const int tid  = threadIdx.x;
  const int lane = tid & 31;
  const int wave = tid >> 5;
  const int m0 = blockIdx.y * 64;
  const int n0 = blockIdx.x * 128;
  const int wm = (wave & 3) * 16;
  const int wn = (wave >> 2) * 64;
  const int l15 = lane & 15;
  const int lhi = lane >> 4;

  // Staging assignments (uint4 = 8 bf16).
  const int arow = tid >> 2;        // 0..63
  const int acg  = (tid & 3) * 8;   // k offset in elements
  const int brow = tid >> 1;        // 0..127
  const int bcg  = (tid & 1) * 16;

  const u16* ah_p = Ahi_g + (size_t)(m0 + arow) * K + acg;
  const u16* al_p = Alo_g + (size_t)(m0 + arow) * K + acg;
  const u16* bh_p = Whi_g + (size_t)(n0 + brow) * K + bcg;
  const u16* bl_p = Wlo_g + (size_t)(n0 + brow) * K + bcg;

  v8f acc[4];
  acc[0] = v8f{}; acc[1] = v8f{}; acc[2] = v8f{}; acc[3] = v8f{};

  uint4 rah, ral, rbh0, rbh1, rbl0, rbl1;
  rah  = *(const uint4*)(ah_p);
  ral  = *(const uint4*)(al_p);
  rbh0 = *(const uint4*)(bh_p);
  rbh1 = *(const uint4*)(bh_p + 8);
  rbl0 = *(const uint4*)(bl_p);
  rbl1 = *(const uint4*)(bl_p + 8);

  for (int k0 = 0; k0 < K; k0 += 32) {
    // ---- stage current chunk to LDS ------------------------------------
    *(uint4*)(&Ahi[arow * RS + acg])     = rah;
    *(uint4*)(&Alo[arow * RS + acg])     = ral;
    *(uint4*)(&Bhi[brow * RS + bcg])     = rbh0;
    *(uint4*)(&Bhi[brow * RS + bcg + 8]) = rbh1;
    *(uint4*)(&Blo[brow * RS + bcg])     = rbl0;
    *(uint4*)(&Blo[brow * RS + bcg + 8]) = rbl1;
    __syncthreads();

    // ---- prefetch next chunk (overlaps WMMA) ---------------------------
    if (k0 + 32 < K) {
      rah  = *(const uint4*)(ah_p + k0 + 32);
      ral  = *(const uint4*)(al_p + k0 + 32);
      rbh0 = *(const uint4*)(bh_p + k0 + 32);
      rbh1 = *(const uint4*)(bh_p + k0 + 40);
      rbl0 = *(const uint4*)(bl_p + k0 + 32);
      rbl1 = *(const uint4*)(bl_p + k0 + 40);
    }

    // ---- compute -------------------------------------------------------
    const int am = (wm + l15) * RS + lhi * 8;
    v16bf ah = ld_frag(&Ahi[am], 16);
    v16bf al = ld_frag(&Alo[am], 16);
#pragma unroll
    for (int j = 0; j < 4; j++) {
      const int bn = (wn + j * 16 + l15) * RS + lhi * 8;
      v16bf bh = ld_frag(&Bhi[bn], 16);
      v16bf bl = ld_frag(&Blo[bn], 16);
      acc[j] = wmma_bf16(al, bh, acc[j]);
      acc[j] = wmma_bf16(ah, bl, acc[j]);
      acc[j] = wmma_bf16(ah, bh, acc[j]);
    }
    __syncthreads();
  }

  // ---- epilogue. C layout: VGPR i, lanes 0-15 -> M=i, 16-31 -> M=i+8. ----
  const int mbase = m0 + wm + 8 * lhi;
#pragma unroll
  for (int i = 0; i < 8; i++) {
    const int m = mbase + i;
#pragma unroll
    for (int j = 0; j < 4; j++) {
      const int n = n0 + wn + j * 16 + l15;
      const float s = acc[j][i] + bias[n];
      const size_t oi = (size_t)m * N + n;
      if (MODE == 0) {
        outF[oi] = s;
      } else if (MODE == 1) {
        const float xa = bf16_to_f32(auxHi[oi]) + bf16_to_f32(auxLo[oi]);
        outF[oi] = xa * sigmoidf_(s);
      } else if (MODE == 2) {
        const float v = siluf_(s) * auxF[oi];
        u16 h, l;
        split1(v, h, l);
        outHi[oi] = h;
        outLo[oi] = l;
      } else {
        outF[oi] = s + auxF[oi];
      }
    }
  }
}

// ---------------------------------------------------------------------------
// Depthwise causal conv1d (k=4) + SiLU; writes xa as split bf16.
// ---------------------------------------------------------------------------
__global__ __launch_bounds__(256) void conv_silu_kernel(
    const float* __restrict__ xp, const float* __restrict__ cw,
    const float* __restrict__ cb, u16* __restrict__ xahi,
    u16* __restrict__ xalo) {
  const int idx = blockIdx.x * 256 + threadIdx.x;  // < NTOK * D_INNER
  const int d = idx % D_INNER;
  const int m = idx / D_INNER;
  const int s = m & (SEQ - 1);
  float acc = cb[d];
#pragma unroll
  for (int j = 0; j < D_CONV; j++) {
    const int sp = s - (D_CONV - 1) + j;
    if (sp >= 0)
      acc += xp[(size_t)(m - (D_CONV - 1) + j) * D_INNER + d] * cw[d * D_CONV + j];
  }
  const float v = siluf_(acc);
  u16 h, l;
  split1(v, h, l);
  xahi[idx] = h;
  xalo[idx] = l;
}

// ---------------------------------------------------------------------------
// Param GEMM (f32 16x16x4 WMMA): decay/Bm/Cm [NTOK,16] from u (f32).
// ---------------------------------------------------------------------------
__global__ __launch_bounds__(256) void param_gemm_kernel(
    const float* __restrict__ u, const float* __restrict__ W_A,
    const float* __restrict__ b_A, const float* __restrict__ W_dt,
    const float* __restrict__ b_dt, const float* __restrict__ W_B,
    const float* __restrict__ b_B, const float* __restrict__ W_C,
    const float* __restrict__ b_C, float* __restrict__ dec,
    float* __restrict__ Bm, float* __restrict__ Cm, int K) {
  const int tid  = threadIdx.x;
  const int lane = tid & 31;
  const int wave = tid >> 5;
  const int m0   = blockIdx.x * 128 + wave * 16;

  v8f aA = {}, aDt = {}, aB = {}, aC = {};
  const int khalf = 2 * (lane >> 4);
  const int l15   = lane & 15;
  const float* ur = u + (size_t)(m0 + l15) * K;

  for (int k = 0; k < K; k += 4) {
    const int kk = k + khalf;
    v2f a, b;
    a[0] = ur[kk];
    a[1] = ur[kk + 1];
    b[0] = W_A[l15 * K + kk];  b[1] = W_A[l15 * K + kk + 1];
    aA = wmma_f32(a, b, aA);
    b[0] = W_dt[l15 * K + kk]; b[1] = W_dt[l15 * K + kk + 1];
    aDt = wmma_f32(a, b, aDt);
    b[0] = W_B[l15 * K + kk];  b[1] = W_B[l15 * K + kk + 1];
    aB = wmma_f32(a, b, aB);
    b[0] = W_C[l15 * K + kk];  b[1] = W_C[l15 * K + kk + 1];
    aC = wmma_f32(a, b, aC);
  }

  const int mb = m0 + 8 * (lane >> 4);
  const int n  = l15;
#pragma unroll
  for (int i = 0; i < 8; i++) {
    const size_t o = (size_t)(mb + i) * D_STATE + n;
    const float Av  = -expf(aA[i] + b_A[n]);
    const float dtv = expf(aDt[i] + b_dt[n]);
    dec[o] = expf(Av * dtv);
    Bm[o]  = aB[i] + b_B[n];
    Cm[o]  = aC[i] + b_C[n];
  }
}

// ---------------------------------------------------------------------------
// Selective scan: 12 blocks = 2 batches x 6 channel-groups of 256.
// ---------------------------------------------------------------------------
__device__ __forceinline__ void load16(const float* __restrict__ p, float* r) {
  const float4* q = (const float4*)p;
  float4 a = q[0], b = q[1], c = q[2], d = q[3];
  r[0] = a.x;  r[1] = a.y;  r[2] = a.z;  r[3] = a.w;
  r[4] = b.x;  r[5] = b.y;  r[6] = b.z;  r[7] = b.w;
  r[8] = c.x;  r[9] = c.y;  r[10] = c.z; r[11] = c.w;
  r[12] = d.x; r[13] = d.y; r[14] = d.z; r[15] = d.w;
}

__global__ __launch_bounds__(256) void scan_kernel(
    const float* __restrict__ u, const float* __restrict__ dec,
    const float* __restrict__ Bm, const float* __restrict__ Cm,
    float* __restrict__ y) {
  const int b = blockIdx.x / (D_INNER / 256);
  const int d = (blockIdx.x % (D_INNER / 256)) * 256 + threadIdx.x;
  float h[D_STATE];
#pragma unroll
  for (int n = 0; n < D_STATE; n++) h[n] = 0.0f;

  const size_t tok0 = (size_t)b * SEQ;
  for (int s = 0; s < SEQ; s++) {
    const size_t t = tok0 + s;
    float dv[16], bv[16], cv[16];
    load16(dec + t * D_STATE, dv);
    load16(Bm + t * D_STATE, bv);
    load16(Cm + t * D_STATE, cv);
    const float uv = u[t * D_INNER + d];
    float yv = 0.0f;
#pragma unroll
    for (int n = 0; n < D_STATE; n++) {
      h[n] = fmaf(h[n], dv[n], bv[n] * uv);
      yv = fmaf(h[n], cv[n], yv);
    }
    y[t * D_INNER + d] = yv;
  }
}

// ---------------------------------------------------------------------------
// Launch
// ---------------------------------------------------------------------------
extern "C" void kernel_launch(void* const* d_in, const int* in_sizes, int n_in,
                              void* d_out, int out_size, void* d_ws,
                              size_t ws_size, hipStream_t stream) {
  const float* x      = (const float*)d_in[0];
  const float* ln_g   = (const float*)d_in[1];
  const float* ln_b   = (const float*)d_in[2];
  const float* W_in   = (const float*)d_in[3];
  const float* b_in   = (const float*)d_in[4];
  const float* conv_w = (const float*)d_in[5];
  const float* conv_b = (const float*)d_in[6];
  const float* W_A    = (const float*)d_in[7];
  const float* b_A    = (const float*)d_in[8];
  const float* W_B    = (const float*)d_in[9];
  const float* b_B    = (const float*)d_in[10];
  const float* W_C    = (const float*)d_in[11];
  const float* b_C    = (const float*)d_in[12];
  const float* W_dt   = (const float*)d_in[13];
  const float* b_dt   = (const float*)d_in[14];
  const float* W_S    = (const float*)d_in[15];  // only first D_INNER rows used
  const float* b_S    = (const float*)d_in[16];
  const float* W_gate = (const float*)d_in[17];
  const float* b_gate = (const float*)d_in[18];
  const float* W_out  = (const float*)d_in[19];
  const float* b_out  = (const float*)d_in[20];
  float* out = (float*)d_out;

  // ---- workspace layout (bytes) -----------------------------------------
  char* base = (char*)d_ws;
  size_t off = 0;
  auto alloc = [&](size_t bytes) {
    char* p = base + off;
    off += (bytes + 255) & ~(size_t)255;
    return p;
  };
  const size_t NM  = (size_t)NTOK * D_MODEL;   // 3,145,728
  const size_t NI  = (size_t)NTOK * D_INNER;   // 6,291,456
  const size_t NS  = (size_t)NTOK * D_STATE;
  const size_t WIN = (size_t)D_INNER * D_MODEL;
  const size_t WSA = (size_t)D_INNER * D_INNER;

  float* xp   = (float*)alloc(NI * 4);   // xp (f32); reused as y after conv
  float* u    = (float*)alloc(NI * 4);
  float* dec  = (float*)alloc(NS * 4);
  float* Bm   = (float*)alloc(NS * 4);
  float* Cm   = (float*)alloc(NS * 4);
  u16* xnhi   = (u16*)alloc(NM * 2);
  u16* xnlo   = (u16*)alloc(NM * 2);
  u16* xahi   = (u16*)alloc(NI * 2);     // reused as g_hi after GEMM(u)
  u16* xalo   = (u16*)alloc(NI * 2);     // reused as g_lo
  u16* wihi   = (u16*)alloc(WIN * 2);
  u16* wilo   = (u16*)alloc(WIN * 2);
  u16* wshi   = (u16*)alloc(WSA * 2);
  u16* wslo   = (u16*)alloc(WSA * 2);
  u16* wghi   = (u16*)alloc(WIN * 2);
  u16* wglo   = (u16*)alloc(WIN * 2);
  u16* wohi   = (u16*)alloc(WIN * 2);
  u16* wolo   = (u16*)alloc(WIN * 2);
  float* y    = xp;
  u16* ghi    = xahi;
  u16* glo    = xalo;

  dim3 blk(256);

  // 0. split weights once (bf16 hi/lo)
  split_kernel<<<dim3(WIN / 1024), blk, 0, stream>>>(W_in, wihi, wilo);
  split_kernel<<<dim3(WSA / 1024), blk, 0, stream>>>(W_S, wshi, wslo);
  split_kernel<<<dim3(WIN / 1024), blk, 0, stream>>>(W_gate, wghi, wglo);
  split_kernel<<<dim3(WIN / 1024), blk, 0, stream>>>(W_out, wohi, wolo);

  // 1. LayerNorm -> split xn
  layernorm_kernel<<<dim3(NTOK), blk, 0, stream>>>(x, ln_g, ln_b, xnhi, xnlo);

  // 2. xp = xn @ W_in.T + b_in   [4096,1536] K=768
  gemm_bf16x3_kernel<0><<<dim3(D_INNER / 128, NTOK / 64), blk, 0, stream>>>(
      xnhi, xnlo, wihi, wilo, b_in, nullptr, nullptr, nullptr, xp, nullptr,
      nullptr, NTOK, D_INNER, D_MODEL);

  // 3. xa = silu(causal_conv(xp)) -> split xa
  conv_silu_kernel<<<dim3((NTOK * D_INNER) / 256), blk, 0, stream>>>(
      xp, conv_w, conv_b, xahi, xalo);

  // 4. u = xa * sigmoid(xa @ W_Sa.T + b_Sa)   K=N=1536
  gemm_bf16x3_kernel<1><<<dim3(D_INNER / 128, NTOK / 64), blk, 0, stream>>>(
      xahi, xalo, wshi, wslo, b_S, nullptr, xahi, xalo, u, nullptr, nullptr,
      NTOK, D_INNER, D_INNER);

  // 5. decay/Bm/Cm from u
  param_gemm_kernel<<<dim3(NTOK / 128), blk, 0, stream>>>(
      u, W_A, b_A, W_dt, b_dt, W_B, b_B, W_C, b_C, dec, Bm, Cm, D_INNER);

  // 6. selective scan -> y  (reuses xp)
  scan_kernel<<<dim3(BATCH * (D_INNER / 256)), blk, 0, stream>>>(u, dec, Bm, Cm, y);

  // 7. g = silu(xn @ W_gate.T + b_gate) * y -> split g  (reuses xa buffers)
  gemm_bf16x3_kernel<2><<<dim3(D_INNER / 128, NTOK / 64), blk, 0, stream>>>(
      xnhi, xnlo, wghi, wglo, b_gate, y, nullptr, nullptr, nullptr, ghi, glo,
      NTOK, D_INNER, D_MODEL);

  // 8. out = g @ W_out.T + b_out + residual   [4096,768] K=1536
  gemm_bf16x3_kernel<3><<<dim3(D_MODEL / 128, NTOK / 64), blk, 0, stream>>>(
      ghi, glo, wohi, wolo, b_out, x, nullptr, nullptr, out, nullptr, nullptr,
      NTOK, D_MODEL, D_INNER);
}
